// ContrastiveLoss_80693845557675
// MI455X (gfx1250) — matrix operand order repaired and checked by reading, not compiled, not run on hardware
//
#include <hip/hip_runtime.h>

#define N_EMB 4096
#define D_EMB 1024
#define MARGIN 0.5f

typedef __bf16 bf16_t;
typedef __attribute__((ext_vector_type(16))) __bf16 v16bf;
typedef __attribute__((ext_vector_type(8)))  __bf16 v8bf;
typedef __attribute__((ext_vector_type(8)))  float  v8f;
typedef __attribute__((ext_vector_type(4)))  float  v4f;

// -----------------------------------------------------------------------------
// Pass 1: one-time fp32 -> bf16 hi/lo split of X (error-compensated operands).
// 4096x1024 elements, 8 per thread, fully coalesced. Output stays in L2 (16 MB).
// -----------------------------------------------------------------------------
__global__ __launch_bounds__(256)
void convert_split_kernel(const float* __restrict__ X,
                          bf16_t* __restrict__ Xhi,
                          bf16_t* __restrict__ Xlo) {
    const size_t base = ((size_t)blockIdx.x * 256 + threadIdx.x) * 8;
    v4f x0 = *(const v4f*)(X + base);
    v4f x1 = *(const v4f*)(X + base + 4);
    v8bf h, l;
#pragma unroll
    for (int t = 0; t < 4; ++t) {
        bf16_t h0 = (bf16_t)x0[t];
        h[t]     = h0;
        l[t]     = (bf16_t)(x0[t] - (float)h0);
        bf16_t h1 = (bf16_t)x1[t];
        h[t + 4] = h1;
        l[t + 4] = (bf16_t)(x1[t] - (float)h1);
    }
    *(v8bf*)(Xhi + base) = h;
    *(v8bf*)(Xlo + base) = l;
}

// -----------------------------------------------------------------------------
// Pass 2: tiled sim = X X^T via WMMA. Each wave: one 16x32 output strip
// (2 accumulators, A reused). Block = 8 row-tiles sharing one column-pair,
// so B operand lines are shared across the whole WGP.
// Inner loop: 12 b128 loads + 6 v_wmma_f32_16x16x32_bf16, no conversions.
// -----------------------------------------------------------------------------
__global__ __launch_bounds__(256)
void contrastive_tile_kernel(const bf16_t* __restrict__ Xhi,
                             const bf16_t* __restrict__ Xlo,
                             const int*    __restrict__ tgt,
                             float*        __restrict__ partial) {
    const int lane = threadIdx.x & 31;
    const int wave = threadIdx.x >> 5;
    const int lg   = lane >> 4;    // lane group (0: lanes 0-15, 1: lanes 16-31)
    const int li   = lane & 15;

    // Block -> (row-tile group, column pair); wave -> row tile within group.
    const int tmg  = blockIdx.x >> 7;      // 0..31
    const int tp   = blockIdx.x & 127;     // 0..127
    const int tm   = tmg * 8 + wave;       // 0..255
    const int row0 = tm * 16;
    const int col0 = tp * 32;

    // A operand (16x32 bf16, MxK), contiguous k order (same as B):
    //   lane group 0: K = k+0..15 ; group 1: K = k+16..31
    const bf16_t* ah_p = Xhi + (size_t)(row0 + li) * D_EMB + lg * 16;
    const bf16_t* al_p = Xlo + (size_t)(row0 + li) * D_EMB + lg * 16;
    // B operand (32x16 bf16, KxN): lane li holds column N=li = X row (col0+li);
    //   lane group 0: K = k+0..15 ; group 1: K = k+16..31 (16 consecutive bf16)
    const bf16_t* b0h_p = Xhi + (size_t)(col0 + li) * D_EMB + lg * 16;
    const bf16_t* b0l_p = Xlo + (size_t)(col0 + li) * D_EMB + lg * 16;
    const bf16_t* b1h_p = Xhi + (size_t)(col0 + 16 + li) * D_EMB + lg * 16;
    const bf16_t* b1l_p = Xlo + (size_t)(col0 + 16 + li) * D_EMB + lg * 16;

    v8f c0 = {};
    v8f c1 = {};

#pragma unroll 2
    for (int k = 0; k < D_EMB; k += 32) {
        v16bf ah, al, b0h, b0l, b1h, b1l;
        // A: two b128 loads per operand (elements 0-7 and 8-15)
        ((v8bf*)&ah)[0] = *(const v8bf*)(ah_p + k);
        ((v8bf*)&ah)[1] = *(const v8bf*)(ah_p + k + 8);
        ((v8bf*)&al)[0] = *(const v8bf*)(al_p + k);
        ((v8bf*)&al)[1] = *(const v8bf*)(al_p + k + 8);
        // B: 16 consecutive bf16 = two b128 loads
        ((v8bf*)&b0h)[0] = *(const v8bf*)(b0h_p + k);
        ((v8bf*)&b0h)[1] = *(const v8bf*)(b0h_p + k + 8);
        ((v8bf*)&b0l)[0] = *(const v8bf*)(b0l_p + k);
        ((v8bf*)&b0l)[1] = *(const v8bf*)(b0l_p + k + 8);
        ((v8bf*)&b1h)[0] = *(const v8bf*)(b1h_p + k);
        ((v8bf*)&b1h)[1] = *(const v8bf*)(b1h_p + k + 8);
        ((v8bf*)&b1l)[0] = *(const v8bf*)(b1l_p + k);
        ((v8bf*)&b1l)[1] = *(const v8bf*)(b1l_p + k + 8);

        // sim += ah*bh + ah*bl + al*bh   (drops only the ~2^-16 lo*lo term)
        c0 = __builtin_amdgcn_wmma_f32_16x16x32_bf16(false, ah, false, b0h, (short)0, c0, false, false);
        c0 = __builtin_amdgcn_wmma_f32_16x16x32_bf16(false, ah, false, b0l, (short)0, c0, false, false);
        c0 = __builtin_amdgcn_wmma_f32_16x16x32_bf16(false, al, false, b0h, (short)0, c0, false, false);
        c1 = __builtin_amdgcn_wmma_f32_16x16x32_bf16(false, ah, false, b1h, (short)0, c1, false, false);
        c1 = __builtin_amdgcn_wmma_f32_16x16x32_bf16(false, ah, false, b1l, (short)0, c1, false, false);
        c1 = __builtin_amdgcn_wmma_f32_16x16x32_bf16(false, al, false, b1h, (short)0, c1, false, false);
    }

    // Elementwise loss on the C/D layout:
    //   lanes 0-15: N = lane, VGPR v -> M = v ; lanes 16-31: N = lane-16, M = v+8
    const int j0  = col0 + li;
    const int j1  = col0 + 16 + li;
    const int tj0 = tgt[j0];
    const int tj1 = tgt[j1];
    const int ib  = row0 + lg * 8;

    float rsum = 0.0f;
#pragma unroll
    for (int v = 0; v < 8; ++v) {
        const int i  = ib + v;
        const int ti = tgt[i];
        const float s0 = c0[v];
        const float s1 = c1[v];
        float t0, t1;
        if (i == j0)        t0 = 0.0f;
        else if (ti == tj0) t0 = 1.0f - s0;
        else                t0 = (s0 > MARGIN) ? s0 : 0.0f;
        if (i == j1)        t1 = 0.0f;
        else if (ti == tj1) t1 = 1.0f - s1;
        else                t1 = (s1 > MARGIN) ? s1 : 0.0f;
        rsum += t0 + t1;
    }

    // Deterministic wave reduce (fixed shuffle order, wave32).
#pragma unroll
    for (int off = 16; off > 0; off >>= 1)
        rsum += __shfl_down(rsum, off, 32);

    __shared__ float smem[8];
    if (lane == 0) smem[wave] = rsum;
    __syncthreads();
    if (threadIdx.x == 0) {
        float bsum = 0.0f;
#pragma unroll
        for (int w = 0; w < 8; ++w) bsum += smem[w];
        partial[blockIdx.x] = bsum;
    }
}

// -----------------------------------------------------------------------------
// Pass 3: single-block deterministic finalize: sum partials, divide by n.
// -----------------------------------------------------------------------------
__global__ __launch_bounds__(256)
void contrastive_finalize_kernel(const float* __restrict__ partial,
                                 int nPartial,
                                 float* __restrict__ out) {
    __shared__ float sm[256];
    float s = 0.0f;
    for (int i = threadIdx.x; i < nPartial; i += 256) s += partial[i];
    sm[threadIdx.x] = s;
    __syncthreads();
    for (int stride = 128; stride > 0; stride >>= 1) {
        if (threadIdx.x < stride) sm[threadIdx.x] += sm[threadIdx.x + stride];
        __syncthreads();
    }
    if (threadIdx.x == 0) out[0] = sm[0] / (float)N_EMB;
}

extern "C" void kernel_launch(void* const* d_in, const int* in_sizes, int n_in,
                              void* d_out, int out_size, void* d_ws, size_t ws_size,
                              hipStream_t stream) {
    const float* X   = (const float*)d_in[0];
    const int*   tgt = (const int*)d_in[1];
    float* out       = (float*)d_out;

    // Workspace layout: Xhi (8 MB) | Xlo (8 MB) | partials (16 KB)
    const size_t elems = (size_t)N_EMB * D_EMB;
    bf16_t* Xhi    = (bf16_t*)d_ws;
    bf16_t* Xlo    = Xhi + elems;
    float*  partial = (float*)(Xlo + elems);

    // Pass 1: split-convert (4096*1024 / (256*8) = 2048 blocks)
    const int convBlocks = (int)(elems / (256 * 8));
    convert_split_kernel<<<convBlocks, 256, 0, stream>>>(X, Xhi, Xlo);

    // Pass 2: WMMA tiles (256 row-tiles x 128 col-pairs, 8 waves/block)
    const int nBlocks = (N_EMB / 16) * (N_EMB / 32) / 8;  // 4096
    contrastive_tile_kernel<<<nBlocks, 256, 0, stream>>>(Xhi, Xlo, tgt, partial);

    // Pass 3: finalize
    contrastive_finalize_kernel<<<1, 256, 0, stream>>>(partial, nBlocks, out);
}
